// SelfSpatialAttentionModule_26499948216896
// MI455X (gfx1250) — matrix operand order repaired and checked by reading, not compiled, not run on hardware
//
#include <hip/hip_runtime.h>
#include <math.h>

typedef __attribute__((ext_vector_type(16))) __bf16 v16bf;
typedef __attribute__((ext_vector_type(8)))  __bf16 v8bf;
typedef __attribute__((ext_vector_type(8)))  float  v8f;

#define BATCH   32
#define CFULL   256
#define CHALF   128
#define HWN     3136     // 56*56
#define WDIM    56
#define MPOOL   784      // 28*28
#define MPAD    800      // pad so K of value-GEMM is a multiple of 32

// -------- WMMA fragment loaders (bf16 16x16x32, contiguous K, same order in A and B) --------
// A (16x32 MxK): lane holds row=lane%16, 16 contiguous K values k0..k0+15,
// k0 = kb (+16 for lanes 16-31). Source must be [M][K] row-major.
static __device__ __forceinline__ v16bf load_a16(const __bf16* p) {
  v8bf lo = *(const v8bf*)(p);        // K = k0 .. k0+7
  v8bf hi = *(const v8bf*)(p + 8);    // K = k0+8 .. k0+15
  return __builtin_shufflevector(lo, hi, 0, 1, 2, 3, 4, 5, 6, 7,
                                         8, 9, 10, 11, 12, 13, 14, 15);
}
// B (32x16 KxN): lane holds column N=lane%16 with 16 *contiguous* K values
// (kb+0..15 for lanes 0-15, kb+16..31 for lanes 16-31). Source must be
// [N][K] so the 16 values are one aligned 32-byte load (2x b128).
static __device__ __forceinline__ v16bf load_b16(const __bf16* p) {
  return *(const v16bf*)(p);
}

static __device__ __forceinline__ v8f wmma_bf16(v16bf a, v16bf b, v8f c) {
  return __builtin_amdgcn_wmma_f32_16x16x32_bf16(false, a, false, b, (short)0, c,
                                                 false, false);
}

// ---------------- prep: f32 -> bf16 conversions ----------------

__global__ void prep_weights_kernel(const float* __restrict__ W1,
                                    const float* __restrict__ W2,
                                    const float* __restrict__ W3,
                                    __bf16* __restrict__ Wcat) {  // [512,256]
  int i = blockIdx.x * blockDim.x + threadIdx.x;
  if (i >= 512 * CFULL) return;
  int o = i / CFULL, c = i % CFULL;
  float v;
  if (o < CHALF)      v = W1[o * CFULL + c];
  else if (o < CFULL) v = W2[(o - CHALF) * CFULL + c];
  else                v = W3[(o - CFULL) * CFULL + c];
  Wcat[i] = (__bf16)v;
}

// x [B,C,HW] f32 -> xbft [B,HW,C] bf16 (transposed so conv B-frags are contiguous)
__global__ void prep_x_kernel(const float* __restrict__ x,
                              __bf16* __restrict__ xbft) {
  int i = blockIdx.x * blockDim.x + threadIdx.x;
  if (i >= BATCH * CFULL * HWN) return;
  int b = i / (CFULL * HWN);
  int r = i % (CFULL * HWN);
  int c = r / HWN, p = r % HWN;
  xbft[((size_t)b * HWN + p) * CFULL + c] = (__bf16)x[i];
}

// ---------------- conv GEMM: Y[512,3136] = Wcat @ x per batch ----------------
// grid (49, 8, B), block 256 = 8 waves (4 in M x 2 in N-groups).
// Block tile 64M x 64N; each wave owns one 16M x 32N strip (2 wmma / K-step).

__global__ void conv_gemm_kernel(const __bf16* __restrict__ Wcat,
                                 const __bf16* __restrict__ xbft,  // [B,3136,256]
                                 __bf16* __restrict__ c1,    // [B,128,3136]
                                 __bf16* __restrict__ c2t,   // [B,3136,128]
                                 __bf16* __restrict__ c3) {  // [B,256,3136]
  const int lane = threadIdx.x & 31;
  const int wave = threadIdx.x >> 5;
  const int wm = wave & 3, wn = wave >> 2;
  const int b  = blockIdx.z;
  const int m0 = blockIdx.y * 64 + wm * 16;
  const int pA = blockIdx.x * 64 + wn * 32 + (lane & 15);       // N col, tile 0
  const int pB = pA + 16;                                       // N col, tile 1
  const __bf16* xb = xbft + (size_t)b * HWN * CFULL;
  const int arow = m0 + (lane & 15);
  const int aoff = (lane < 16) ? 0 : 16;
  const int boff = (lane < 16) ? 0 : 16;

  v8f acc0 = {}, acc1 = {};
#pragma unroll
  for (int ks = 0; ks < 8; ++ks) {
    const int kb = ks * 32;
    if (ks < 7)
      __builtin_prefetch(&xb[(size_t)pA * CFULL + kb + 32], 0, 3);
    v16bf a  = load_a16(Wcat + (size_t)arow * CFULL + kb + aoff);
    v16bf b0 = load_b16(xb + (size_t)pA * CFULL + kb + boff);
    v16bf b1 = load_b16(xb + (size_t)pB * CFULL + kb + boff);
    acc0 = wmma_bf16(a, b0, acc0);
    acc1 = wmma_bf16(a, b1, acc1);
  }

#pragma unroll
  for (int r = 0; r < 8; ++r) {
    const int mrow = m0 + r + ((lane >= 16) ? 8 : 0);
    const __bf16 v0 = (__bf16)acc0[r];
    const __bf16 v1 = (__bf16)acc1[r];
    if (mrow < CHALF) {
      c1[((size_t)b * CHALF + mrow) * HWN + pA] = v0;
      c1[((size_t)b * CHALF + mrow) * HWN + pB] = v1;
    } else if (mrow < CFULL) {
      // c2 stored transposed [n,128]: feeds WMMA-A row-major in attention
      c2t[((size_t)b * HWN + pA) * CHALF + (mrow - CHALF)] = v0;
      c2t[((size_t)b * HWN + pB) * CHALF + (mrow - CHALF)] = v1;
    } else {
      c3[((size_t)b * CFULL + (mrow - CFULL)) * HWN + pA] = v0;
      c3[((size_t)b * CFULL + (mrow - CFULL)) * HWN + pB] = v1;
    }
  }
}

// ---------------- 2x2 max+avg pooling ----------------

static __device__ __forceinline__ float pool4(const __bf16* __restrict__ src,
                                              size_t chanBase, int h2, int w2) {
  size_t base = chanBase + (size_t)(2 * h2) * WDIM + 2 * w2;
  float a = (float)src[base],        bb = (float)src[base + 1];
  float c = (float)src[base + WDIM], d  = (float)src[base + WDIM + 1];
  float mx = fmaxf(fmaxf(a, bb), fmaxf(c, d));
  return mx + 0.25f * (a + bb + c + d);
}

// c1 [B,128,3136] -> c1pt [B,784,128]  (transposed: score-GEMM B-frags contiguous)
__global__ void pool_c1_kernel(const __bf16* __restrict__ c1,
                               __bf16* __restrict__ c1pt) {
  int i = blockIdx.x * blockDim.x + threadIdx.x;
  if (i >= BATCH * MPOOL * CHALF) return;
  int b = i / (MPOOL * CHALF);
  int r = i % (MPOOL * CHALF);
  int m = r / CHALF, c = r % CHALF;
  c1pt[i] = (__bf16)pool4(c1, ((size_t)b * CHALF + c) * HWN, m / 28, m % 28);
}

// c3 [B,256,3136] -> c3p [B,256,800]  (m fastest, zero-padded to 800)
__global__ void pool_c3_kernel(const __bf16* __restrict__ c3,
                               __bf16* __restrict__ c3p) {
  int i = blockIdx.x * blockDim.x + threadIdx.x;
  if (i >= BATCH * CFULL * MPAD) return;
  int b = i / (CFULL * MPAD);
  int r = i % (CFULL * MPAD);
  int c = r / MPAD, m = r % MPAD;
  float v = 0.f;
  if (m < MPOOL)
    v = pool4(c3, ((size_t)b * CFULL + c) * HWN, m / 28, m % 28);
  c3p[i] = (__bf16)v;  // pad columns 784..799 are zero => clean 25x32 K-loop
}

// ---------------- fused attention: scores -> softmax -> value GEMM ----------------
// grid (196, B), block 256 (8 waves). One block owns 16 rows of n.
// Dynamic LDS: f32 scores [16][800] (51.2 KB) + bf16 probs [16][800] (25.6 KB).

#define SMEM_BYTES (16 * MPAD * 4 + 16 * MPAD * 2)

__global__ void attention_kernel(const __bf16* __restrict__ c2t,   // [B,3136,128]
                                 const __bf16* __restrict__ c1pt,  // [B,784,128]
                                 const __bf16* __restrict__ c3p,   // [B,256,800]
                                 const float*  __restrict__ x,
                                 const float*  __restrict__ gamma,
                                 float* __restrict__ out) {
  extern __shared__ char smem[];
  float*  s_sc = (float*)smem;                        // [16][MPAD] f32 scores
  __bf16* s_pb = (__bf16*)(smem + 16 * MPAD * 4);     // [16][MPAD] bf16 probs

  const int lane = threadIdx.x & 31;
  const int wave = threadIdx.x >> 5;
  const int b  = blockIdx.y;
  const int n0 = blockIdx.x * 16;
  const int aoff = (lane < 16) ? 0 : 16;
  const int boff = (lane < 16) ? 0 : 16;

  const __bf16* c2b  = c2t  + (size_t)b * HWN * CHALF;
  const __bf16* c1pb = c1pt + (size_t)b * MPOOL * CHALF;
  const __bf16* c3b  = c3p  + (size_t)b * CFULL * MPAD;

  // Preload the 16x128 A panel (this n-tile of c2^T) once: 4 fragments.
  const int arow = n0 + (lane & 15);
  v16bf afr[4];
#pragma unroll
  for (int ks = 0; ks < 4; ++ks)
    afr[ks] = load_a16(c2b + (size_t)arow * CHALF + ks * 32 + aoff);

  // --- score GEMM: s[16,784] = A(16x128) @ c1p(128x784); m-tiles round-robin ---
  for (int mt = wave; mt < 49; mt += 8) {
    v8f acc = {};
    const int mcol = mt * 16 + (lane & 15);
    const __bf16* brow = c1pb + (size_t)mcol * CHALF + boff;
#pragma unroll
    for (int ks = 0; ks < 4; ++ks)
      acc = wmma_bf16(afr[ks], load_b16(brow + ks * 32), acc);
#pragma unroll
    for (int r = 0; r < 8; ++r)
      s_sc[(r + ((lane >= 16) ? 8 : 0)) * MPAD + mcol] = acc[r];
  }
  __syncthreads();

  // --- softmax over m (784) per row; wave w owns rows 2w, 2w+1.
  //     Emits bf16 probabilities directly: value-GEMM A-frags need no cvt. ---
#pragma unroll
  for (int rr = 0; rr < 2; ++rr) {
    const int row = wave * 2 + rr;
    float*  sr = s_sc + (size_t)row * MPAD;
    __bf16* pr = s_pb + (size_t)row * MPAD;
    float mx = -INFINITY;
    for (int m = lane; m < MPOOL; m += 32) mx = fmaxf(mx, sr[m]);
#pragma unroll
    for (int off = 16; off > 0; off >>= 1) mx = fmaxf(mx, __shfl_xor(mx, off, 32));
    float sum = 0.f;
    for (int m = lane; m < MPOOL; m += 32) {
      float e = __expf(sr[m] - mx);
      sr[m] = e;
      sum += e;
    }
#pragma unroll
    for (int off = 16; off > 0; off >>= 1) sum += __shfl_xor(sum, off, 32);
    const float inv = 1.0f / sum;
    for (int m = lane; m < MPOOL; m += 32) pr[m] = (__bf16)(sr[m] * inv);
    for (int m = MPOOL + lane; m < MPAD; m += 32) pr[m] = (__bf16)0.f;  // pad
  }
  __syncthreads();

  const float g = gamma[0];

  // --- value GEMM: D[16,256] = P(16x800) @ c3p^T; wave owns c-tiles w and w+8.
  //     A-frag = 2x ds_read_b128 of native bf16 probs; B-frag = 32B global. ---
  const __bf16* prow = s_pb + (size_t)(lane & 15) * MPAD;
  const int cA = wave * 16 + (lane & 15);
  const int cB = (wave + 8) * 16 + (lane & 15);
  v8f acc0 = {}, acc1 = {};
#pragma unroll 1
  for (int ks = 0; ks < 25; ++ks) {
    const int kb = ks * 32;
    v16bf a  = load_a16(prow + kb + aoff);
    v16bf b0 = load_b16(c3b + (size_t)cA * MPAD + kb + boff);
    v16bf b1 = load_b16(c3b + (size_t)cB * MPAD + kb + boff);
    acc0 = wmma_bf16(a, b0, acc0);   // shared A fragment across both tiles
    acc1 = wmma_bf16(a, b1, acc1);
  }

  // --- fused epilogue: out = gamma * refined + x ---
#pragma unroll
  for (int r = 0; r < 8; ++r) {
    const int n = n0 + r + ((lane >= 16) ? 8 : 0);
    const size_t iA = ((size_t)b * CFULL + cA) * HWN + n;
    const size_t iB = ((size_t)b * CFULL + cB) * HWN + n;
    out[iA] = g * acc0[r] + x[iA];
    out[iB] = g * acc1[r] + x[iB];
  }
}

// ---------------- launch ----------------

extern "C" void kernel_launch(void* const* d_in, const int* in_sizes, int n_in,
                              void* d_out, int out_size, void* d_ws, size_t ws_size,
                              hipStream_t stream) {
  (void)in_sizes; (void)n_in; (void)out_size; (void)ws_size;
  const float* x  = (const float*)d_in[0];
  const float* W1 = (const float*)d_in[1];
  const float* W2 = (const float*)d_in[2];
  const float* W3 = (const float*)d_in[3];
  const float* gm = (const float*)d_in[4];
  float* out = (float*)d_out;

  // workspace carve-up (bf16 intermediates), 256B aligned
  char* ws = (char*)d_ws;
  size_t off = 0;
  auto carve = [&](size_t bytes) {
    char* p = ws + off;
    off += (bytes + 255) & ~(size_t)255;
    return p;
  };
  __bf16* Wcat = (__bf16*)carve((size_t)512 * CFULL * 2);
  __bf16* xbft = (__bf16*)carve((size_t)BATCH * HWN * CFULL * 2);
  __bf16* c1   = (__bf16*)carve((size_t)BATCH * CHALF * HWN * 2);
  __bf16* c2t  = (__bf16*)carve((size_t)BATCH * HWN * CHALF * 2);
  __bf16* c3   = (__bf16*)carve((size_t)BATCH * CFULL * HWN * 2);
  __bf16* c1pt = (__bf16*)carve((size_t)BATCH * MPOOL * CHALF * 2);
  __bf16* c3p  = (__bf16*)carve((size_t)BATCH * CFULL * MPAD * 2);

  const int xelems = BATCH * CFULL * HWN;

  prep_weights_kernel<<<(512 * CFULL + 255) / 256, 256, 0, stream>>>(W1, W2, W3, Wcat);
  prep_x_kernel<<<(xelems + 255) / 256, 256, 0, stream>>>(x, xbft);
  conv_gemm_kernel<<<dim3(HWN / 64, 512 / 64, BATCH), 256, 0, stream>>>(
      Wcat, xbft, c1, c2t, c3);
  pool_c1_kernel<<<(BATCH * MPOOL * CHALF + 255) / 256, 256, 0, stream>>>(c1, c1pt);
  pool_c3_kernel<<<(BATCH * CFULL * MPAD + 255) / 256, 256, 0, stream>>>(c3, c3p);
  attention_kernel<<<dim3(HWN / 16, BATCH), 256, SMEM_BYTES, stream>>>(
      c2t, c1pt, c3p, x, gm, out);
}
